// GAT_69389491634486
// MI455X (gfx1250) — matrix operand order repaired and checked by reading, not compiled, not run on hardware
//
#include <hip/hip_runtime.h>
#include <hip/hip_bf16.h>

typedef __attribute__((ext_vector_type(16))) __bf16 bf16x16;
typedef __attribute__((ext_vector_type(8)))  float  f32x8;

#define NEG_SLOPE 0.2f
#define EPS_F 1e-16f

// ---------- order-preserving float<->uint key for atomicMax on floats ----------
__device__ __forceinline__ unsigned fkey(float x) {
    unsigned b = __float_as_uint(x);
    return (b & 0x80000000u) ? ~b : (b | 0x80000000u);
}
__device__ __forceinline__ float funkey(unsigned k) {
    unsigned b = (k & 0x80000000u) ? (k & 0x7fffffffu) : ~k;
    return __uint_as_float(b);
}

// ---------- prep: transpose weights to bf16 [K, 64] ----------
__global__ void prep_weights(const float* __restrict__ W1, const float* __restrict__ W2,
                             __bf16* __restrict__ W1t, __bf16* __restrict__ W2t) {
    int t = blockIdx.x * blockDim.x + threadIdx.x;
    if (t < 256 * 64) {
        int k = t >> 6, j = t & 63;
        W1t[t] = (__bf16)W1[j * 256 + k];          // W1: [64,256] -> W1t: [256,64]
    }
    if (t < 64 * 64) {
        int k = t >> 6, j = t & 63;
        W2t[t] = (__bf16)W2[j * 64 + k];           // W2: [64,64]  -> W2t: [64,64]
    }
}

// ---------- GEMM1: h1[N,64] = x[N,256] @ W1^T + b1  (bf16 WMMA, f32 accum) ----------
__global__ __launch_bounds__(256) void gemm1_kernel(const float* __restrict__ x,
                                                    const __bf16* __restrict__ Wt,
                                                    const float* __restrict__ bias,
                                                    float* __restrict__ h, int N) {
    const int wave = threadIdx.x >> 5;
    const int lane = threadIdx.x & 31;
    const int mtile = blockIdx.x * 8 + wave;
    if (mtile * 16 >= N) return;                    // uniform per wave: EXEC stays all-ones
    const int row = mtile * 16 + (lane & 15);
    const int kb  = (lane >> 4) * 8;
    const float* xr = x + (size_t)row * 256;

    f32x8 acc[4] = {};
    for (int ks = 0; ks < 256; ks += 32) {
        const float* p0 = xr + ks + kb;             // K = ks+kb .. +7
        const float* p1 = xr + ks + 16 + kb;        // K = ks+16+kb .. +7
        bf16x16 a;
        #pragma unroll
        for (int i = 0; i < 8; ++i) { a[i] = (__bf16)p0[i]; a[8 + i] = (__bf16)p1[i]; }
        const __bf16* wrow = Wt + (size_t)(ks + kb) * 64 + (lane & 15);   // B: lane holds column n = lane&15, K in table order
        #pragma unroll
        for (int t = 0; t < 4; ++t) {
            bf16x16 b;
            #pragma unroll
            for (int i = 0; i < 8; ++i) { b[i] = wrow[i * 64 + t * 16]; b[8 + i] = wrow[(16 + i) * 64 + t * 16]; }
            acc[t] = __builtin_amdgcn_wmma_f32_16x16x32_bf16(false, a, false, b,
                                                             (short)0, acc[t], false, false);
        }
    }
    const int mrow0 = mtile * 16 + 8 * (lane >> 4);
    const int ncol  = lane & 15;
    #pragma unroll
    for (int t = 0; t < 4; ++t)
        #pragma unroll
        for (int r = 0; r < 8; ++r) {
            int j = t * 16 + ncol;
            h[(size_t)(mrow0 + r) * 64 + j] = acc[t][r] + bias[j];
        }
}

// ---------- GEMM2: h2[N,64] = relu(out1+b1) @ W2^T  (ReLU fused into A conversion) ----------
__global__ __launch_bounds__(256) void gemm2_kernel(const float* __restrict__ act,
                                                    const float* __restrict__ b1,
                                                    const __bf16* __restrict__ Wt,
                                                    float* __restrict__ h, int N) {
    const int wave = threadIdx.x >> 5;
    const int lane = threadIdx.x & 31;
    const int mtile = blockIdx.x * 8 + wave;
    if (mtile * 16 >= N) return;
    const int row = mtile * 16 + (lane & 15);
    const int kb  = (lane >> 4) * 8;
    const float* xr = act + (size_t)row * 64;

    f32x8 acc[4] = {};
    for (int ks = 0; ks < 64; ks += 32) {
        bf16x16 a;
        #pragma unroll
        for (int i = 0; i < 8; ++i) {
            int k0 = ks + kb + i, k1 = ks + 16 + kb + i;
            a[i]     = (__bf16)fmaxf(xr[k0] + b1[k0], 0.f);
            a[8 + i] = (__bf16)fmaxf(xr[k1] + b1[k1], 0.f);
        }
        const __bf16* wrow = Wt + (size_t)(ks + kb) * 64 + (lane & 15);
        #pragma unroll
        for (int t = 0; t < 4; ++t) {
            bf16x16 b;
            #pragma unroll
            for (int i = 0; i < 8; ++i) { b[i] = wrow[i * 64 + t * 16]; b[8 + i] = wrow[(16 + i) * 64 + t * 16]; }
            acc[t] = __builtin_amdgcn_wmma_f32_16x16x32_bf16(false, a, false, b,
                                                             (short)0, acc[t], false, false);
        }
    }
    const int mrow0 = mtile * 16 + 8 * (lane >> 4);
    const int ncol  = lane & 15;
    #pragma unroll
    for (int t = 0; t < 4; ++t)
        #pragma unroll
        for (int r = 0; r < 8; ++r)
            h[(size_t)(mrow0 + r) * 64 + t * 16 + ncol] = acc[t][r];
}

// ---------- attention projections ----------
__global__ void attn1_kernel(const float* __restrict__ h1, const float* __restrict__ asrc,
                             const float* __restrict__ adst, float* __restrict__ as,
                             float* __restrict__ ad, int N) {
    int t = blockIdx.x * blockDim.x + threadIdx.x;
    if (t >= N * 8) return;
    int n = t >> 3, hh = t & 7;
    const float* hp = h1 + (size_t)n * 64 + hh * 8;
    float s = 0.f, d = 0.f;
    #pragma unroll
    for (int i = 0; i < 8; ++i) { float v = hp[i]; s += v * asrc[hh * 8 + i]; d += v * adst[hh * 8 + i]; }
    as[t] = s; ad[t] = d;
}

__global__ void attn2_kernel(const float* __restrict__ h2, const float* __restrict__ asrc,
                             const float* __restrict__ adst, float* __restrict__ as,
                             float* __restrict__ ad, int N) {
    int n = blockIdx.x * blockDim.x + threadIdx.x;
    if (n >= N) return;
    const float* hp = h2 + (size_t)n * 64;
    float s = 0.f, d = 0.f;
    for (int i = 0; i < 64; ++i) { float v = hp[i]; s += v * asrc[i]; d += v * adst[i]; }
    as[n] = s; ad[n] = d;
}

// ---------- layer-1 edge passes (8 heads) ----------
__device__ __forceinline__ void edge_sd(int e, int E, const int* src, const int* dst, int& s, int& d) {
    if (e < E) { s = src[e]; d = dst[e]; } else { s = d = e - E; }   // appended self-loops
}

__global__ void edge_max1(const int* __restrict__ src, const int* __restrict__ dst, int E, int ET8,
                          const float* __restrict__ as, const float* __restrict__ ad,
                          unsigned* __restrict__ mkey) {
    int t = blockIdx.x * blockDim.x + threadIdx.x;
    if (t >= ET8) return;
    int e = t >> 3, h = t & 7, s, d;
    edge_sd(e, E, src, dst, s, d);
    float v = as[s * 8 + h] + ad[d * 8 + h];
    v = v > 0.f ? v : NEG_SLOPE * v;
    atomicMax(&mkey[d * 8 + h], fkey(v));
}

__global__ void edge_sum1(const int* __restrict__ src, const int* __restrict__ dst, int E, int ET8,
                          const float* __restrict__ as, const float* __restrict__ ad,
                          const unsigned* __restrict__ mkey, float* __restrict__ ssum) {
    int t = blockIdx.x * blockDim.x + threadIdx.x;
    if (t >= ET8) return;
    int e = t >> 3, h = t & 7, s, d;
    edge_sd(e, E, src, dst, s, d);
    float v = as[s * 8 + h] + ad[d * 8 + h];
    v = v > 0.f ? v : NEG_SLOPE * v;
    atomicAdd(&ssum[d * 8 + h], expf(v - funkey(mkey[d * 8 + h])));
}

__global__ void edge_agg1(const int* __restrict__ src, const int* __restrict__ dst, int E, int ET8,
                          const float* __restrict__ as, const float* __restrict__ ad,
                          const unsigned* __restrict__ mkey, const float* __restrict__ ssum,
                          const float* __restrict__ h1, float* __restrict__ out1) {
    int t = blockIdx.x * blockDim.x + threadIdx.x;
    if (t >= ET8) return;
    int e = t >> 3, h = t & 7, s, d;
    edge_sd(e, E, src, dst, s, d);
    float v = as[s * 8 + h] + ad[d * 8 + h];
    v = v > 0.f ? v : NEG_SLOPE * v;
    float alpha = expf(v - funkey(mkey[d * 8 + h])) / (ssum[d * 8 + h] + EPS_F);
    const float* hs = h1 + (size_t)s * 64 + h * 8;
    float* od = out1 + (size_t)d * 64 + h * 8;
    #pragma unroll
    for (int i = 0; i < 8; ++i) atomicAdd(od + i, hs[i] * alpha);
}

// ---------- layer-2 edge passes (1 head, 64 dims) ----------
__global__ void edge_max2(const int* __restrict__ src, const int* __restrict__ dst, int E, int ET,
                          const float* __restrict__ as, const float* __restrict__ ad,
                          unsigned* __restrict__ mkey) {
    int e = blockIdx.x * blockDim.x + threadIdx.x;
    if (e >= ET) return;
    int s, d; edge_sd(e, E, src, dst, s, d);
    float v = as[s] + ad[d];
    v = v > 0.f ? v : NEG_SLOPE * v;
    atomicMax(&mkey[d], fkey(v));
}

__global__ void edge_sum2(const int* __restrict__ src, const int* __restrict__ dst, int E, int ET,
                          const float* __restrict__ as, const float* __restrict__ ad,
                          const unsigned* __restrict__ mkey, float* __restrict__ ssum) {
    int e = blockIdx.x * blockDim.x + threadIdx.x;
    if (e >= ET) return;
    int s, d; edge_sd(e, E, src, dst, s, d);
    float v = as[s] + ad[d];
    v = v > 0.f ? v : NEG_SLOPE * v;
    atomicAdd(&ssum[d], expf(v - funkey(mkey[d])));
}

__global__ void edge_agg2(const int* __restrict__ src, const int* __restrict__ dst, int E, int ET8,
                          const float* __restrict__ as, const float* __restrict__ ad,
                          const unsigned* __restrict__ mkey, const float* __restrict__ ssum,
                          const float* __restrict__ h2, float* __restrict__ out2) {
    int t = blockIdx.x * blockDim.x + threadIdx.x;
    if (t >= ET8) return;
    int e = t >> 3, part = t & 7, s, d;
    edge_sd(e, E, src, dst, s, d);
    float v = as[s] + ad[d];
    v = v > 0.f ? v : NEG_SLOPE * v;
    float alpha = expf(v - funkey(mkey[d])) / (ssum[d] + EPS_F);
    const float* hs = h2 + (size_t)s * 64 + part * 8;
    float* od = out2 + (size_t)d * 64 + part * 8;
    #pragma unroll
    for (int i = 0; i < 8; ++i) atomicAdd(od + i, hs[i] * alpha);
}

// ---------- final: log_softmax(out2 + b2), one wave per node ----------
__global__ __launch_bounds__(256) void lsm_kernel(const float* __restrict__ out2,
                                                  const float* __restrict__ b2,
                                                  float* __restrict__ y, int N) {
    int wave = threadIdx.x >> 5, lane = threadIdx.x & 31;
    int n = blockIdx.x * 8 + wave;
    if (n >= N) return;
    const float* p = out2 + (size_t)n * 64;
    float v0 = p[lane] + b2[lane];
    float v1 = p[lane + 32] + b2[lane + 32];
    float mx = fmaxf(v0, v1);
    #pragma unroll
    for (int off = 16; off; off >>= 1) mx = fmaxf(mx, __shfl_xor(mx, off, 32));
    float se = expf(v0 - mx) + expf(v1 - mx);
    #pragma unroll
    for (int off = 16; off; off >>= 1) se += __shfl_xor(se, off, 32);
    float lse = logf(se) + mx;
    y[(size_t)n * 64 + lane]      = v0 - lse;
    y[(size_t)n * 64 + lane + 32] = v1 - lse;
}

extern "C" void kernel_launch(void* const* d_in, const int* in_sizes, int n_in,
                              void* d_out, int out_size, void* d_ws, size_t ws_size,
                              hipStream_t stream) {
    const float* x        = (const float*)d_in[0];
    const int*   src      = (const int*)d_in[1];
    const float* W1       = (const float*)d_in[2];
    const float* att_src1 = (const float*)d_in[3];
    const float* att_dst1 = (const float*)d_in[4];
    const float* b1       = (const float*)d_in[5];
    const float* W2       = (const float*)d_in[6];
    const float* att_src2 = (const float*)d_in[7];
    const float* att_dst2 = (const float*)d_in[8];
    const float* b2       = (const float*)d_in[9];
    float* y = (float*)d_out;

    const int N  = in_sizes[0] / 256;
    const int E  = in_sizes[1] / 2;
    const int* dst = src + E;
    const int ET  = E + N;            // with self-loops
    const int ET8 = ET * 8;

    // workspace carve-out (256B aligned slots)
    char* base = (char*)d_ws;
    size_t off = 0;
    auto carve = [&](size_t bytes) { void* p = base + off; off += (bytes + 255) & ~(size_t)255; return p; };
    // zero-initialized contiguous span first:
    unsigned* m1k  = (unsigned*)carve((size_t)N * 8 * 4);
    float*    s1   = (float*)   carve((size_t)N * 8 * 4);
    float*    out1 = (float*)   carve((size_t)N * 64 * 4);
    unsigned* m2k  = (unsigned*)carve((size_t)N * 4);
    float*    s2   = (float*)   carve((size_t)N * 4);
    float*    out2 = (float*)   carve((size_t)N * 64 * 4);
    size_t zero_bytes = off;
    // write-before-read buffers:
    float*  h1  = (float*) carve((size_t)N * 64 * 4);
    float*  h2  = (float*) carve((size_t)N * 64 * 4);
    float*  as1 = (float*) carve((size_t)N * 8 * 4);
    float*  ad1 = (float*) carve((size_t)N * 8 * 4);
    float*  as2 = (float*) carve((size_t)N * 4);
    float*  ad2 = (float*) carve((size_t)N * 4);
    __bf16* W1t = (__bf16*)carve((size_t)256 * 64 * 2);
    __bf16* W2t = (__bf16*)carve((size_t)64 * 64 * 2);

    hipMemsetAsync(d_ws, 0, zero_bytes, stream);
    prep_weights<<<(256 * 64 + 255) / 256, 256, 0, stream>>>(W1, W2, W1t, W2t);

    // ---- layer 1 ----
    gemm1_kernel<<<(N + 127) / 128, 256, 0, stream>>>(x, W1t, b1, h1, N);
    attn1_kernel<<<(N * 8 + 255) / 256, 256, 0, stream>>>(h1, att_src1, att_dst1, as1, ad1, N);
    edge_max1<<<(ET8 + 255) / 256, 256, 0, stream>>>(src, dst, E, ET8, as1, ad1, m1k);
    edge_sum1<<<(ET8 + 255) / 256, 256, 0, stream>>>(src, dst, E, ET8, as1, ad1, m1k, s1);
    edge_agg1<<<(ET8 + 255) / 256, 256, 0, stream>>>(src, dst, E, ET8, as1, ad1, m1k, s1, h1, out1);

    // ---- layer 2 (relu(out1+b1) fused into GEMM2 A-side) ----
    gemm2_kernel<<<(N + 127) / 128, 256, 0, stream>>>(out1, b1, W2t, h2, N);
    attn2_kernel<<<(N + 255) / 256, 256, 0, stream>>>(h2, att_src2, att_dst2, as2, ad2, N);
    edge_max2<<<(ET + 255) / 256, 256, 0, stream>>>(src, dst, E, ET, as2, ad2, m2k);
    edge_sum2<<<(ET + 255) / 256, 256, 0, stream>>>(src, dst, E, ET, as2, ad2, m2k, s2);
    edge_agg2<<<(ET8 + 255) / 256, 256, 0, stream>>>(src, dst, E, ET8, as2, ad2, m2k, s2, h2, out2);

    // ---- output ----
    lsm_kernel<<<(N + 7) / 8, 256, 0, stream>>>(out2, b2, y, N);
}
